// GCN_1881195676180
// MI455X (gfx1250) — matrix operand order repaired and checked by reading, not compiled, not run on hardware
//
#include <hip/hip_runtime.h>

typedef __attribute__((ext_vector_type(16))) _Float16 v16h;
typedef __attribute__((ext_vector_type(8)))  float    v8f;

#define F_IN 128
#define H1DIM 16
#define H2DIM 64
#define H3DIM 128

// ---------------- graph normalization ----------------
__global__ void k_init_deg(float* deg, int n) {
  int i = blockIdx.x * blockDim.x + threadIdx.x;
  if (i < n) deg[i] = 1.0f;                      // self loop contributes 1
}

__global__ void k_count_deg(const long long* __restrict__ dst, float* deg, int e) {
  int i = blockIdx.x * blockDim.x + threadIdx.x;
  if (i < e) atomicAdd(&deg[(int)dst[i]], 1.0f);
}

__global__ void k_rsqrt(float* deg, int n) {
  int i = blockIdx.x * blockDim.x + threadIdx.x;
  if (i < n) deg[i] = rsqrtf(deg[i]);            // in place: deg -> dinv
}

// precompute int32 indices + per-edge symmetric norm (used by all 3 layers)
__global__ void k_edge_prep(const long long* __restrict__ src, const long long* __restrict__ dst,
                            const float* __restrict__ dinv, int* __restrict__ src32,
                            int* __restrict__ dst32, float* __restrict__ norm, int e) {
  int i = blockIdx.x * blockDim.x + threadIdx.x;
  if (i >= e) return;
  int s = (int)src[i], d = (int)dst[i];
  src32[i] = s; dst32[i] = d;
  norm[i] = dinv[s] * dinv[d];
}

// ---------------- pack W[K,O] into WMMA B-fragment layout ----------------
// B fragment (32x16 f16), lane=(g,n): element e -> W[k0 + (e<8 ? 8g+e : 16+8g+e-8)][col0 + n]
// Packed layout: P[((ct*KC + kc)*32 + lane)*16 + e]
template<int K, int O>
__global__ void k_pack_w(const float* __restrict__ W, _Float16* __restrict__ P) {
  constexpr int KC = (K + 31) / 32;
  const int lane = threadIdx.x;
  const int g = lane >> 4, n = lane & 15;
  const int ct = blockIdx.x, kc = blockIdx.y;
  v16h b;
#pragma unroll
  for (int e = 0; e < 16; ++e) {
    int kk = kc * 32 + ((e < 8) ? (8 * g + e) : (16 + 8 * g + (e - 8)));
    b[e] = (_Float16)((kk < K) ? W[(size_t)kk * O + ct * 16 + n] : 0.0f);
  }
  *(reinterpret_cast<v16h*>(P) + (size_t)(ct * KC + kc) * 32 + lane) = b;
}

// ---------------- dense transform: D[N,O] = act(A[N,K]) @ W[K,O] ----------------
// One wave per 16x16 tile. A fragment built from 2x b128 loads per half; B fragment
// is one 32B vector load from the pre-packed buffer. Fully unrolled -> KC wmmas.
template<bool RELU>
__device__ inline void cvt8(const float* __restrict__ p, v16h& a, int base) {
  float4 v0 = reinterpret_cast<const float4*>(p)[0];
  float4 v1 = reinterpret_cast<const float4*>(p)[1];
  if (RELU) {
    v0.x = fmaxf(v0.x, 0.f); v0.y = fmaxf(v0.y, 0.f); v0.z = fmaxf(v0.z, 0.f); v0.w = fmaxf(v0.w, 0.f);
    v1.x = fmaxf(v1.x, 0.f); v1.y = fmaxf(v1.y, 0.f); v1.z = fmaxf(v1.z, 0.f); v1.w = fmaxf(v1.w, 0.f);
  }
  a[base + 0] = (_Float16)v0.x; a[base + 1] = (_Float16)v0.y;
  a[base + 2] = (_Float16)v0.z; a[base + 3] = (_Float16)v0.w;
  a[base + 4] = (_Float16)v1.x; a[base + 5] = (_Float16)v1.y;
  a[base + 6] = (_Float16)v1.z; a[base + 7] = (_Float16)v1.w;
}

template<int K, int O, bool RELU_IN>
__global__ void k_gemm_wmma(const float* __restrict__ A, const _Float16* __restrict__ WP,
                            float* __restrict__ D) {
  constexpr int KC = (K + 31) / 32;
  const int lane = threadIdx.x;              // 0..31, no divergence anywhere
  const int g = lane >> 4, m = lane & 15;
  const int row0 = blockIdx.x << 4;
  const int ct = blockIdx.y;
  const float* __restrict__ arow = A + (size_t)(row0 + m) * K;
  const v16h* __restrict__ bfrag =
      reinterpret_cast<const v16h*>(WP) + (size_t)ct * KC * 32 + lane;

  v8f acc = {};
#pragma unroll
  for (int kc = 0; kc < KC; ++kc) {
    const int k0 = kc * 32;
    v16h a;
    cvt8<RELU_IN>(arow + k0 + 8 * g, a, 0);            // K = k0+8g .. +7 (always in range)
    if constexpr (K % 32 == 0) {
      cvt8<RELU_IN>(arow + k0 + 16 + 8 * g, a, 8);     // K = k0+16+8g .. +7
    } else {                                           // K=16 layer: upper half is zero
#pragma unroll
      for (int e = 8; e < 16; ++e) a[e] = (_Float16)0.f;
    }
    v16h b = bfrag[kc * 32];
    acc = __builtin_amdgcn_wmma_f32_16x16x32_f16(
        false, a, false, b, (short)0, acc, false, false);
  }
#pragma unroll
  for (int r = 0; r < 8; ++r)
    D[(size_t)(row0 + 8 * g + r) * O + ct * 16 + m] = acc[r];
}

// ---------------- aggregation ----------------
// init: out = self-loop message + bias  (vectorized float4)
template<int O>
__global__ void k_self_bias(const float* __restrict__ t, const float* __restrict__ dinv,
                            const float* __restrict__ bias, float* __restrict__ out, int n) {
  constexpr int OQ = O / 4;
  int idx = blockIdx.x * blockDim.x + threadIdx.x;
  if (idx >= n * OQ) return;
  int i = idx / OQ, q = idx % OQ;            // OQ is a power-of-two constant
  float d = dinv[i], dd = d * d;
  float4 tv = reinterpret_cast<const float4*>(t)[idx];
  float4 bv = reinterpret_cast<const float4*>(bias)[q];
  float4 o{tv.x * dd + bv.x, tv.y * dd + bv.y, tv.z * dd + bv.z, tv.w * dd + bv.w};
  reinterpret_cast<float4*>(out)[idx] = o;
}

// edge scatter: thread handles one edge x 4 features
template<int O>
__global__ void k_scatter(const float* __restrict__ t, const int* __restrict__ src,
                          const int* __restrict__ dst, const float* __restrict__ norm,
                          float* __restrict__ out, int E) {
  constexpr int OQ = O / 4;
  long long idx = (long long)blockIdx.x * blockDim.x + threadIdx.x;
  if (idx >= (long long)E * OQ) return;
  int e = (int)(idx / OQ), q = (int)(idx % OQ);
  int s = src[e], d = dst[e];
  float w = norm[e];
  float4 tv = reinterpret_cast<const float4*>(t + (size_t)s * O)[q];
  float* o = out + (size_t)d * O + 4 * q;
  atomicAdd(o + 0, tv.x * w);
  atomicAdd(o + 1, tv.y * w);
  atomicAdd(o + 2, tv.z * w);
  atomicAdd(o + 3, tv.w * w);
}

// ---------------- final FC: out[i] = sum_f relu(h3[i][f]) * Wfc[f] + bfc ----------------
__global__ void k_fc(const float* __restrict__ H, const float* __restrict__ Wfc,
                     const float* __restrict__ bfc, float* __restrict__ out, int n) {
  int wid  = (int)(((long long)blockIdx.x * blockDim.x + threadIdx.x) >> 5);
  int lane = threadIdx.x & 31;
  if (wid >= n) return;
  const float* h = H + (size_t)wid * H3DIM;
  float s = 0.0f;
#pragma unroll
  for (int f = lane; f < H3DIM; f += 32)
    s += fmaxf(h[f], 0.0f) * Wfc[f];
#pragma unroll
  for (int off = 16; off > 0; off >>= 1)
    s += __shfl_down(s, off, 32);
  if (lane == 0) out[wid] = s + bfc[0];
}

extern "C" void kernel_launch(void* const* d_in, const int* in_sizes, int n_in,
                              void* d_out, int out_size, void* d_ws, size_t ws_size,
                              hipStream_t stream) {
  const float*     x    = (const float*)d_in[0];
  const long long* eidx = (const long long*)d_in[1];   // int64 (2,E)
  const float*     W1   = (const float*)d_in[2];
  const float*     b1   = (const float*)d_in[3];
  const float*     W2   = (const float*)d_in[4];
  const float*     b2   = (const float*)d_in[5];
  const float*     W3   = (const float*)d_in[6];
  const float*     b3   = (const float*)d_in[7];
  const float*     Wfc  = (const float*)d_in[8];
  const float*     bfc  = (const float*)d_in[9];

  const int N = in_sizes[0] / F_IN;   // 100000 (divisible by 16)
  const int E = in_sizes[1] / 2;      // 1.6M
  const long long* srcL = eidx;
  const long long* dstL = eidx + E;

  // 128B-aligned bump allocator over d_ws
  char* wp = (char*)d_ws;
  auto alloc = [&](size_t bytes) {
    void* r = (void*)wp;
    wp += (bytes + 127) & ~(size_t)127;
    return r;
  };
  float*    dinv  = (float*)alloc((size_t)N * 4);
  float*    bufA  = (float*)alloc((size_t)N * 128 * 4);
  float*    bufB  = (float*)alloc((size_t)N * 128 * 4);
  int*      src32 = (int*)  alloc((size_t)E * 4);
  int*      dst32 = (int*)  alloc((size_t)E * 4);
  float*    norm  = (float*)alloc((size_t)E * 4);
  _Float16* WP1   = (_Float16*)alloc((size_t)(H1DIM/16) * 4 * 32 * 16 * 2); // KC=4
  _Float16* WP2   = (_Float16*)alloc((size_t)(H2DIM/16) * 1 * 32 * 16 * 2); // KC=1
  _Float16* WP3   = (_Float16*)alloc((size_t)(H3DIM/16) * 2 * 32 * 16 * 2); // KC=2
  float* out = (float*)d_out;

  const int B = 256;

  // graph normalization coefficients + edge prep
  k_init_deg <<<(N + B - 1) / B, B, 0, stream>>>(dinv, N);
  k_count_deg<<<(E + B - 1) / B, B, 0, stream>>>(dstL, dinv, E);
  k_rsqrt    <<<(N + B - 1) / B, B, 0, stream>>>(dinv, N);
  k_edge_prep<<<(E + B - 1) / B, B, 0, stream>>>(srcL, dstL, dinv, src32, dst32, norm, E);

  // pack weights into B-fragment layout (tiny)
  k_pack_w<F_IN,  H1DIM><<<dim3(H1DIM/16, 4), 32, 0, stream>>>(W1, WP1);
  k_pack_w<H1DIM, H2DIM><<<dim3(H2DIM/16, 1), 32, 0, stream>>>(W2, WP2);
  k_pack_w<H2DIM, H3DIM><<<dim3(H3DIM/16, 2), 32, 0, stream>>>(W3, WP3);

  // layer 1: x[N,128] @ W1 -> t1[N,16]; aggregate -> a1 (relu deferred to consumer)
  k_gemm_wmma<F_IN, H1DIM, false><<<dim3(N/16, H1DIM/16), 32, 0, stream>>>(x, WP1, bufA);
  k_self_bias<H1DIM><<<((N * (H1DIM/4)) + B - 1) / B, B, 0, stream>>>(bufA, dinv, b1, bufB, N);
  k_scatter<H1DIM><<<(unsigned)(((long long)E * (H1DIM/4) + B - 1) / B), B, 0, stream>>>(
      bufA, src32, dst32, norm, bufB, E);

  // layer 2: relu(a1)[N,16] @ W2 -> t2[N,64]; aggregate -> a2
  k_gemm_wmma<H1DIM, H2DIM, true><<<dim3(N/16, H2DIM/16), 32, 0, stream>>>(bufB, WP2, bufA);
  k_self_bias<H2DIM><<<((N * (H2DIM/4)) + B - 1) / B, B, 0, stream>>>(bufA, dinv, b2, bufB, N);
  k_scatter<H2DIM><<<(unsigned)(((long long)E * (H2DIM/4) + B - 1) / B), B, 0, stream>>>(
      bufA, src32, dst32, norm, bufB, E);

  // layer 3: relu(a2)[N,64] @ W3 -> t3[N,128]; aggregate -> a3
  k_gemm_wmma<H2DIM, H3DIM, true><<<dim3(N/16, H3DIM/16), 32, 0, stream>>>(bufB, WP3, bufA);
  k_self_bias<H3DIM><<<((N * (H3DIM/4)) + B - 1) / B, B, 0, stream>>>(bufA, dinv, b3, bufB, N);
  k_scatter<H3DIM><<<(unsigned)(((long long)E * (H3DIM/4) + B - 1) / B), B, 0, stream>>>(
      bufA, src32, dst32, norm, bufB, E);

  // FC head: one wave per node, ReLU fused
  k_fc<<<(N + 7) / 8, 256, 0, stream>>>(bufB, Wfc, bfc, out, N);
}
